// MLA_66391604461900
// MI455X (gfx1250) — compile-verified, matrix operand order checked by reading, not run on hardware
//
#include <hip/hip_runtime.h>
#include <hip/hip_bf16.h>

// ---------------- problem constants (from reference) ----------------
constexpr int CB    = 2;
constexpr int CT    = 2048;
constexpr int CC    = 2048;
constexpr int CH    = 16;
constexpr int CQL   = 1536;     // Q_LORA
constexpr int CROPE = 64;
constexpr int CNOPE = 128;
constexpr int CQHD  = CNOPE + CROPE;   // 192
constexpr int CKVL  = 512;
constexpr int CVHD  = 128;
constexpr int CMT   = CB * CT;  // 4096 rows
constexpr float CSCALE = 0.07216878364870323f;            // 192^-0.5
constexpr float CSCALE2 = CSCALE * 1.4426950408889634f;   // SCALE * log2(e)
constexpr float CEPS   = 1e-6f;

typedef unsigned short u16;
typedef __attribute__((ext_vector_type(4)))  unsigned int u32x4;
typedef __attribute__((ext_vector_type(8)))  unsigned int u32x8;
typedef __attribute__((ext_vector_type(8)))  float v8f;
typedef __attribute__((ext_vector_type(8)))  unsigned short v8u16;
typedef __attribute__((ext_vector_type(16))) unsigned short v16u16;
typedef __attribute__((ext_vector_type(16))) __bf16 v16bf;

// round-to-nearest-even f32 -> bf16 (bit pattern)
__device__ __forceinline__ u16 f2bf(float f) {
    unsigned u = __builtin_bit_cast(unsigned, f);
    u += 0x7FFFu + ((u >> 16) & 1u);
    return (u16)(u >> 16);
}

// Load one lane's 16 bf16 halves of a 16x32 WMMA A/B fragment.
// Per the CDNA5 ISA 16-bit layout, lane (m = l%16, h = l/16) holds
// K = {8h..8h+7} (vector idx 0..7) and K = {16+8h..16+8h+7} (idx 8..15),
// i.e. two contiguous 16-byte runs at p and p+16 elements.
__device__ __forceinline__ v16bf load_frag(const u16* p) {
    v8u16 lo = *(const v8u16*)(p);
    v8u16 hi = *(const v8u16*)(p + 16);
    v16u16 u = __builtin_shufflevector(lo, hi,
        0,1,2,3,4,5,6,7,8,9,10,11,12,13,14,15);
    return __builtin_bit_cast(v16bf, u);
}

__device__ __forceinline__ v8f wmma_bf16(v16bf a, v16bf b, v8f c) {
    return __builtin_amdgcn_wmma_f32_16x16x32_bf16(
        false, a, false, b, (short)0, c, false, false);
}

// ---------------- Tensor Data Mover: 2-D tile global -> LDS ----------------
// D# per CDNA5 ISA ch.8: group0 = {flags, lds_addr, global_addr, type},
// group1 = {mask/data_size/pad, tensor dims, tile dims, dim0 stride}.
// Tile: rows x 32 bf16 elements, row stride = row_stride_elems (in elements).
// pad_enable: after every 16 DWORDs (one 64B row) insert 4 DWORDs -> LDS row
// stride of 40 u16 (bank-conflict-free fragment reads).
__device__ __forceinline__ void tdm_load_tile(const u16* gsrc,
                                              unsigned lds_byte_addr,
                                              unsigned rows,
                                              unsigned long long row_stride_elems) {
    unsigned long long ga = (unsigned long long)(uintptr_t)gsrc;
    u32x4 g0;
    g0.x = 1u;                                   // count=1, user descriptor
    g0.y = lds_byte_addr;                        // LDS byte address
    g0.z = (unsigned)(ga & 0xFFFFFFFFu);         // global addr [31:0]
    g0.w = (unsigned)((ga >> 32) & 0x01FFFFFFu)  // global addr [56:32]
         | (2u << 30);                           // type = 2 (image/tensor)
    const unsigned td0 = 32u;                    // tensor/tile dim0 (elements)
    const unsigned td1 = rows;
    u32x8 g1;
    g1.s0 = (1u << 16)        // data_size = 1 -> 2 bytes
          | (1u << 20)        // pad_enable
          | (3u << 22)        // pad_interval = 3 -> 16 DWORDs
          | (3u << 25);       // pad_amount = 3 -> 4 DWORDs
    g1.s1 = (td0 & 0xFFFFu) << 16;                       // tensor_dim0 lo16
    g1.s2 = (td0 >> 16) | ((td1 & 0xFFFFu) << 16);       // dim0 hi / dim1 lo
    g1.s3 = (td1 >> 16) | (32u << 16);                   // dim1 hi / tile_dim0
    g1.s4 = rows & 0xFFFFu;                              // tile_dim1 (tile_dim2=0)
    g1.s5 = (unsigned)(row_stride_elems & 0xFFFFFFFFull);
    g1.s6 = (unsigned)((row_stride_elems >> 32) & 0xFFFFull);  // stride1 = 0
    g1.s7 = 0u;
    asm volatile("tensor_load_to_lds %0, %1" :: "s"(g0), "s"(g1) : "memory");
}

// ---------------- elementwise f32 -> bf16 ----------------
__global__ __launch_bounds__(256) void cvt_f32_bf16(const float* __restrict__ in,
                                                    u16* __restrict__ out,
                                                    size_t n) {
    size_t i = (size_t)blockIdx.x * 256 + threadIdx.x;
    if (i < n) out[i] = f2bf(in[i]);
}

// ---------------- TDM-staged NT bf16 WMMA GEMM ----------------
// C[M,N] = A[M,K] * B[N,K]^T. Block = 256 threads (8 waves), block tile
// 128 x BN, K-chunk 32, double-buffered LDS tiles filled by the TDM.
// Wave (mw,nw) computes a 32 x (BN/2) sub-tile: 8 (BN=128) or 4 (BN=64)
// WMMAs per chunk from LDS fragments (row stride 40 u16 via TDM padding).
template<int BN, int BF16OUT>
__global__ __launch_bounds__(256) void gemm_tdm(const u16* __restrict__ A,
                                                const u16* __restrict__ B,
                                                void* __restrict__ Cout,
                                                int M, int N, int K) {
    constexpr int NJ = BN / 32;          // per-wave N tiles of 16
    __shared__ u16 As[2][128 * 40];
    __shared__ u16 Bs[2][BN * 40];

    const int wid  = threadIdx.x >> 5;
    const int lane = threadIdx.x & 31;
    const int nn = lane & 15;
    const int hh = lane >> 4;
    const int mw = wid & 3;              // 4 M-groups of 32 rows
    const int nw = wid >> 2;             // 2 N-groups of BN/2 cols
    const int bm0 = blockIdx.y * 128;
    const int bn0 = blockIdx.x * BN;

    v8f acc[2][NJ];
    #pragma unroll
    for (int s = 0; s < 2; ++s)
        #pragma unroll
        for (int j = 0; j < NJ; ++j)
            #pragma unroll
            for (int e = 0; e < 8; ++e) acc[s][j][e] = 0.0f;

    const int nchunks = K >> 5;
    if (wid == 0) {
        tdm_load_tile(A + (size_t)bm0 * K, (unsigned)(uintptr_t)&As[0][0],
                      128u, (unsigned long long)K);
        tdm_load_tile(B + (size_t)bn0 * K, (unsigned)(uintptr_t)&Bs[0][0],
                      (unsigned)BN, (unsigned long long)K);
    }

    for (int c = 0; c < nchunks; ++c) {
        const int cur = c & 1;
        if (wid == 0) {
            if (c + 1 < nchunks) {
                const int k1 = (c + 1) << 5;
                tdm_load_tile(A + (size_t)bm0 * K + k1,
                              (unsigned)(uintptr_t)&As[cur ^ 1][0],
                              128u, (unsigned long long)K);
                tdm_load_tile(B + (size_t)bn0 * K + k1,
                              (unsigned)(uintptr_t)&Bs[cur ^ 1][0],
                              (unsigned)BN, (unsigned long long)K);
                __builtin_amdgcn_s_wait_tensorcnt(2);   // current tiles done
            } else {
                __builtin_amdgcn_s_wait_tensorcnt(0);
            }
        }
        __syncthreads();

        v16bf a0 = load_frag(&As[cur][(mw * 32 + nn) * 40 + 8 * hh]);
        v16bf a1 = load_frag(&As[cur][(mw * 32 + 16 + nn) * 40 + 8 * hh]);
        #pragma unroll
        for (int j = 0; j < NJ; ++j) {
            v16bf b = load_frag(&Bs[cur][(nw * (BN / 2) + j * 16 + nn) * 40 + 8 * hh]);
            acc[0][j] = wmma_bf16(a0, b, acc[0][j]);
            acc[1][j] = wmma_bf16(a1, b, acc[1][j]);
        }
        __syncthreads();
    }

    // D layout: row = v + 8*(lane/16), col = lane%16
    #pragma unroll
    for (int s = 0; s < 2; ++s)
        #pragma unroll
        for (int j = 0; j < NJ; ++j)
            #pragma unroll
            for (int v = 0; v < 8; ++v) {
                size_t idx = (size_t)(bm0 + mw * 32 + s * 16 + v + 8 * hh) * N
                           + bn0 + nw * (BN / 2) + j * 16 + nn;
                float val = acc[s][j][v];
                if (BF16OUT) ((u16*)Cout)[idx] = f2bf(val);
                else         ((float*)Cout)[idx] = val;
            }
}

// ---------------- rmsnorm (f32 in, bf16 out); optional k_pe slice ----------------
__global__ __launch_bounds__(256) void rmsnorm_kernel(const float* __restrict__ in,
                                                      const float* __restrict__ w,
                                                      u16* __restrict__ out,
                                                      u16* __restrict__ kpe,
                                                      int width, int stride_in) {
    const int row = blockIdx.x;
    const float* x = in + (size_t)row * stride_in;
    __shared__ float red[8];
    __shared__ float invs;

    float ss = 0.0f;
    for (int i = threadIdx.x; i < width; i += 256) {
        float v = x[i];
        ss += v * v;
    }
    #pragma unroll
    for (int off = 16; off; off >>= 1) ss += __shfl_xor(ss, off, 32);
    if ((threadIdx.x & 31) == 0) red[threadIdx.x >> 5] = ss;
    __syncthreads();
    if (threadIdx.x == 0) {
        float t = 0.0f;
        #pragma unroll
        for (int i = 0; i < 8; ++i) t += red[i];
        invs = rsqrtf(t * (1.0f / width) + CEPS);
    }
    __syncthreads();
    const float s = invs;
    for (int i = threadIdx.x; i < width; i += 256)
        out[(size_t)row * width + i] = f2bf(w[i] * x[i] * s);
    if (kpe != nullptr && threadIdx.x < CROPE)
        kpe[(size_t)row * CROPE + threadIdx.x] = f2bf(x[width + threadIdx.x]);
}

// ---------------- build V^T[b,h,d,t] bf16 from kv bf16 ----------------
__global__ __launch_bounds__(256) void build_vT(const u16* __restrict__ kvb,
                                                u16* __restrict__ vT) {
    size_t i = (size_t)blockIdx.x * 256 + threadIdx.x;
    size_t total = (size_t)CB * CH * CVHD * CT;
    if (i >= total) return;
    int t = (int)(i % CT); size_t r = i / CT;
    int d = (int)(r % CVHD); size_t r2 = r / CVHD;
    int h = (int)(r2 % CH);
    int b = (int)(r2 / CH);
    vT[(((size_t)(b * CH + h) * CVHD + d) * CT) + t] =
        kvb[(size_t)(b * CT + t) * (CH * (CNOPE + CVHD)) + h * (CNOPE + CVHD) + CNOPE + d];
}

// ---------------- fused flash attention (one wave per 16-row q tile) ----------------
// q_b:  [B*T, H*192] bf16 ; kvb: [B*T, H*256] bf16 (k_nope at +0, v at +128)
// kpe:  [B*T, 64] bf16    ; vT : [B,H,128,T] bf16
// attn: [B*T, H*128] bf16
// q tile is staged in LDS (stride 200 u16 -> 100 DWORDs, conflict-free) and
// fragments are re-materialized per K-chunk to keep VGPR count < 256.
__global__ __launch_bounds__(32) void mla_attn(const u16* __restrict__ qb,
                                               const u16* __restrict__ kvb,
                                               const u16* __restrict__ kpe,
                                               const u16* __restrict__ vT,
                                               u16* __restrict__ attn) {
    constexpr int QT = CT / 16;        // 128 q tiles per (b,h)
    constexpr int KVW = CH * (CNOPE + CVHD); // 4096
    constexpr int QS = 200;            // LDS q row stride (u16)
    const int gid = blockIdx.x;
    const int qt = gid % QT;
    const int h  = (gid / QT) % CH;
    const int b  = gid / (QT * CH);
    const int lane = threadIdx.x;
    const int nn = lane & 15;
    const int hh = lane >> 4;
    const int qlast = qt * 16 + 15;
    const float NEG_INF = -__builtin_inff();

    __shared__ u16 Qs[16 * QS];   // 16 q rows x 192 dims (padded stride)
    __shared__ u16 Ps[16 * 40];   // 16 q rows x 32 key cols (stride 40, padded)

    // cooperative stage of the q tile: 16 rows x 24 chunks of 8 u16
    {
        const u16* qbase = qb + (size_t)(b * CT + qt * 16) * (CH * CQHD) + h * CQHD;
        for (int idx = lane; idx < 16 * 24; idx += 32) {
            int row = idx / 24, ch = idx % 24;
            *(v8u16*)(&Qs[row * QS + ch * 8]) =
                *(const v8u16*)(qbase + (size_t)row * (CH * CQHD) + ch * 8);
        }
    }
    __syncthreads();

    v8f o[8];
    float mrow[8], lrow[8];
    #pragma unroll
    for (int v = 0; v < 8; ++v) { mrow[v] = NEG_INF; lrow[v] = 0.0f; }
    #pragma unroll
    for (int j = 0; j < 8; ++j)
        #pragma unroll
        for (int e = 0; e < 8; ++e) o[j][e] = 0.0f;

    for (int kt = 0; kt <= qlast; kt += 32) {
        v8f s0, s1;
        #pragma unroll
        for (int e = 0; e < 8; ++e) { s0[e] = 0.0f; s1[e] = 0.0f; }

        const size_t rk0 = (size_t)(b * CT) + kt + nn;       // key row for s0 (col nn)
        const size_t rk1 = rk0 + 16;                         // key row for s1
        #pragma unroll
        for (int c = 0; c < 6; ++c) {
            const int d0 = c * 32;
            const u16 *p0, *p1;
            if (d0 < CNOPE) {      // k_nope from kv
                p0 = kvb + rk0 * KVW + h * (CNOPE + CVHD) + d0 + 8 * hh;
                p1 = kvb + rk1 * KVW + h * (CNOPE + CVHD) + d0 + 8 * hh;
            } else {               // k_pe (shared across heads)
                p0 = kpe + rk0 * CROPE + (d0 - CNOPE) + 8 * hh;
                p1 = kpe + rk1 * CROPE + (d0 - CNOPE) + 8 * hh;
            }
            v16bf qc = load_frag(&Qs[nn * QS + d0 + 8 * hh]);   // A-frag from LDS
            s0 = wmma_bf16(qc, load_frag(p0), s0);
            s1 = wmma_bf16(qc, load_frag(p1), s1);
        }

        // online softmax (base-2) over the 32 new key columns
        #pragma unroll
        for (int v = 0; v < 8; ++v) {
            const int rowg = qt * 16 + v + 8 * hh;
            float e0 = s0[v] * CSCALE2;     // logits * SCALE * log2(e)
            float e1 = s1[v] * CSCALE2;
            if (kt + nn      > rowg) e0 = NEG_INF;   // causal
            if (kt + 16 + nn > rowg) e1 = NEG_INF;
            float mx = fmaxf(e0, e1);
            #pragma unroll
            for (int off = 8; off; off >>= 1) mx = fmaxf(mx, __shfl_xor(mx, off, 32));
            const float mnew  = fmaxf(mrow[v], mx);
            const float alpha = exp2f(mrow[v] - mnew);
            mrow[v] = mnew;
            const float p0 = exp2f(e0 - mnew);
            const float p1 = exp2f(e1 - mnew);
            float rs = p0 + p1;
            #pragma unroll
            for (int off = 8; off; off >>= 1) rs += __shfl_xor(rs, off, 32);
            lrow[v] = lrow[v] * alpha + rs;
            #pragma unroll
            for (int j = 0; j < 8; ++j) o[j][v] *= alpha;
            Ps[(v + 8 * hh) * 40 + nn]      = f2bf(p0);
            Ps[(v + 8 * hh) * 40 + 16 + nn] = f2bf(p1);
        }
        __syncthreads();

        // P (16x32) @ V (32x128) via 8 WMMAs; A-frag from LDS (M = nn)
        v16bf aP = load_frag(&Ps[nn * 40 + 8 * hh]);
        #pragma unroll
        for (int j = 0; j < 8; ++j) {
            const u16* vp = vT + ((size_t)(b * CH + h) * CVHD + j * 16 + nn) * CT
                               + kt + 8 * hh;
            o[j] = wmma_bf16(aP, load_frag(vp), o[j]);
        }
        __syncthreads();
    }

    // normalize + store, layout [b,t, h*128+d]
    #pragma unroll
    for (int v = 0; v < 8; ++v) {
        const float inv = 1.0f / lrow[v];
        const size_t rowo = (size_t)(b * CT + qt * 16 + v + 8 * hh);
        #pragma unroll
        for (int j = 0; j < 8; ++j)
            attn[rowo * (CH * CVHD) + h * CVHD + j * 16 + nn] = f2bf(o[j][v] * inv);
    }
}

// ---------------- host launcher ----------------
extern "C" void kernel_launch(void* const* d_in, const int* in_sizes, int n_in,
                              void* d_out, int out_size, void* d_ws, size_t ws_size,
                              hipStream_t stream) {
    (void)in_sizes; (void)n_in; (void)out_size; (void)ws_size;
    const float* x    = (const float*)d_in[0];
    const float* Wqa  = (const float*)d_in[1];
    const float* gqa  = (const float*)d_in[2];
    const float* Wqb  = (const float*)d_in[3];
    const float* Wkva = (const float*)d_in[4];
    const float* gkva = (const float*)d_in[5];
    const float* Wkvb = (const float*)d_in[6];
    const float* Wo   = (const float*)d_in[7];
    float* out = (float*)d_out;

    char* ws = (char*)d_ws;
    size_t off = 0;
    auto alloc = [&](size_t bytes) -> char* {
        off = (off + 255) & ~(size_t)255;
        char* p = ws + off;
        off += bytes;
        return p;
    };

    constexpr size_t N_X    = (size_t)CMT * CC;
    constexpr size_t N_WQA  = (size_t)CQL * CC;
    constexpr size_t N_WQB  = (size_t)(CH * CQHD) * CQL;
    constexpr size_t N_WKVA = (size_t)(CKVL + CROPE) * CC;
    constexpr size_t N_WKVB = (size_t)(CH * (CNOPE + CVHD)) * CKVL;
    constexpr size_t N_WO   = (size_t)CC * (CH * CVHD);

    u16*   x_b     = (u16*)alloc(N_X * 2);
    u16*   Wqa_b   = (u16*)alloc(N_WQA * 2);
    u16*   Wqb_b   = (u16*)alloc(N_WQB * 2);
    u16*   Wkva_b  = (u16*)alloc(N_WKVA * 2);
    u16*   Wkvb_b  = (u16*)alloc(N_WKVB * 2);
    u16*   Wo_b    = (u16*)alloc(N_WO * 2);
    float* qlora_f = (float*)alloc((size_t)CMT * CQL * 4);
    u16*   qlora_n = (u16*)alloc((size_t)CMT * CQL * 2);
    u16*   q_b     = (u16*)alloc((size_t)CMT * CH * CQHD * 2);
    float* ckv_f   = (float*)alloc((size_t)CMT * (CKVL + CROPE) * 4);
    u16*   ckv_n   = (u16*)alloc((size_t)CMT * CKVL * 2);
    u16*   kpe_b   = (u16*)alloc((size_t)CMT * CROPE * 2);
    u16*   kv_b    = (u16*)alloc((size_t)CMT * CH * (CNOPE + CVHD) * 2);
    u16*   vT_b    = (u16*)alloc((size_t)CB * CH * CVHD * CT * 2);
    u16*   attn_b  = (u16*)alloc((size_t)CMT * CH * CVHD * 2);

    auto cvt = [&](const float* src, u16* dst, size_t n) {
        cvt_f32_bf16<<<dim3((unsigned)((n + 255) / 256)), dim3(256), 0, stream>>>(src, dst, n);
    };
    cvt(x,    x_b,    N_X);
    cvt(Wqa,  Wqa_b,  N_WQA);
    cvt(Wqb,  Wqb_b,  N_WQB);
    cvt(Wkva, Wkva_b, N_WKVA);
    cvt(Wkvb, Wkvb_b, N_WKVB);
    cvt(Wo,   Wo_b,   N_WO);

    // q_lora = x @ Wqa^T   [4096 x 1536] f32
    gemm_tdm<128, 0><<<dim3(CQL / 128, CMT / 128), dim3(256), 0, stream>>>(
        x_b, Wqa_b, qlora_f, CMT, CQL, CC);
    // ckv = x @ Wkva^T     [4096 x 576] f32 (N%128!=0 -> BN=64)
    gemm_tdm<64, 0><<<dim3((CKVL + CROPE) / 64, CMT / 128), dim3(256), 0, stream>>>(
        x_b, Wkva_b, ckv_f, CMT, CKVL + CROPE, CC);

    // rmsnorms (+ k_pe slice-out)
    rmsnorm_kernel<<<dim3(CMT), dim3(256), 0, stream>>>(
        qlora_f, gqa, qlora_n, nullptr, CQL, CQL);
    rmsnorm_kernel<<<dim3(CMT), dim3(256), 0, stream>>>(
        ckv_f, gkva, ckv_n, kpe_b, CKVL, CKVL + CROPE);

    // q = rms(q_lora) @ Wqb^T  [4096 x 3072] bf16
    gemm_tdm<128, 1><<<dim3((CH * CQHD) / 128, CMT / 128), dim3(256), 0, stream>>>(
        qlora_n, Wqb_b, q_b, CMT, CH * CQHD, CQL);
    // kv = rms(ckv) @ Wkvb^T   [4096 x 4096] bf16
    gemm_tdm<128, 1><<<dim3((CH * (CNOPE + CVHD)) / 128, CMT / 128), dim3(256), 0, stream>>>(
        ckv_n, Wkvb_b, kv_b, CMT, CH * (CNOPE + CVHD), CKVL);

    // V^T for the PV matmul
    {
        size_t total = (size_t)CB * CH * CVHD * CT;
        build_vT<<<dim3((unsigned)((total + 255) / 256)), dim3(256), 0, stream>>>(kv_b, vT_b);
    }

    // fused flash attention: one wave per (b, h, 16-row q tile)
    mla_attn<<<dim3(CB * CH * (CT / 16)), dim3(32), 0, stream>>>(
        q_b, kv_b, kpe_b, vT_b, attn_b);

    // out = attn @ Wo^T  [4096 x 2048] f32
    gemm_tdm<128, 0><<<dim3(CC / 128, CMT / 128), dim3(256), 0, stream>>>(
        attn_b, Wo_b, out, CMT, CC, CH * CVHD);
}
